// TFXLNetRelativeAttention_64665027609150
// MI455X (gfx1250) — matrix operand order repaired and checked by reading, not compiled, not run on hardware
//
#include <hip/hip_runtime.h>
#include <hip/hip_bf16.h>

typedef __attribute__((ext_vector_type(16))) _Float16 v16h;
typedef __attribute__((ext_vector_type(8)))  _Float16 v8h;
typedef __attribute__((ext_vector_type(8)))  float    v8f;

#define QLEN 1024
#define MLEN 1024
#define BSZ  2
#define DM   1024
#define NH   16
#define DH   64
#define KLEN 2048
#define RLEN 3072
#define SCALE 0.125f
#define LN_EPS 1e-12f

// ---------------------------------------------------------------------------
// WMMA helpers (CDNA5 v_wmma_f32_16x16x32_f16, wave32)
// A-frag (16x32 f16): lane half h=lane/16, row m=lane%16.
//   elems 0..7  = A[m, K0 + 16h + 0..7]
//   elems 8..15 = A[m, K0 + 16h + 8 + 0..7]
// B-frag (32x16 f16): col n=lane%16, elems e = B[K0 + 16h + e, n]  (buffer [col][k])
// C/D (16x16 f32): col n=lane%16, reg r holds row m = r + 8h
// ---------------------------------------------------------------------------
__device__ __forceinline__ v8f wmma32(v16h a, v16h b, v8f c) {
  return __builtin_amdgcn_wmma_f32_16x16x32_f16(false, a, false, b, (short)0, c,
                                                false, false);
}

__device__ __forceinline__ v16h load_afrag(const _Float16* __restrict__ row,
                                           int K0, int half) {
  v8h lo = *(const v8h*)(row + K0 + 16 * half);
  v8h hi = *(const v8h*)(row + K0 + 16 * half + 8);
  v16h r;
#pragma unroll
  for (int e = 0; e < 8; ++e) { r[e] = lo[e]; r[e + 8] = hi[e]; }
  return r;
}

// ---------------------------------------------------------------------------
// Conversion / staging kernels
// ---------------------------------------------------------------------------
__global__ void conv_cat(const float* __restrict__ mems, const float* __restrict__ h,
                         _Float16* __restrict__ cat16) {
  size_t i = (size_t)blockIdx.x * 256 + threadIdx.x;
  const size_t memsz = (size_t)MLEN * BSZ * DM;
  const size_t tot   = (size_t)KLEN * BSZ * DM;
  if (i < tot) cat16[i] = (_Float16)(i < memsz ? mems[i] : h[i - memsz]);
}

__global__ void conv_f16(const float* __restrict__ src, _Float16* __restrict__ dst,
                         size_t n) {
  size_t i = (size_t)blockIdx.x * 256 + threadIdx.x;
  if (i < n) dst[i] = (_Float16)src[i];
}

// W[k][c] (f32) -> Wt[c][k] (f16)   ([col][k] layout for B-fragments)
__global__ void conv_wT(const float* __restrict__ W, _Float16* __restrict__ Wt) {
  size_t i = (size_t)blockIdx.x * 256 + threadIdx.x;
  if (i < (size_t)DM * DM) {
    int k = (int)(i / DM), c = (int)(i % DM);
    Wt[(size_t)c * DM + k] = (_Float16)W[i];
  }
}

// sd8[b][j][i] = 1 if seg differs (seg_mat[i,j,b,1] != 0)
__global__ void sd8_kernel(const float* __restrict__ seg_mat,
                           unsigned char* __restrict__ sd8) {
  size_t t = (size_t)blockIdx.x * 256 + threadIdx.x;
  if (t >= (size_t)BSZ * KLEN * QLEN) return;
  int i = (int)(t % QLEN);
  int j = (int)((t / QLEN) % KLEN);
  int b = (int)(t / ((size_t)QLEN * KLEN));
  float s1 = seg_mat[(((size_t)i * KLEN + j) * BSZ + b) * 2 + 1];
  sd8[t] = (s1 != 0.f) ? 1 : 0;
}

// ---------------------------------------------------------------------------
// Projection GEMM, 16x64 output per wave (A-fragment reused across 4 N-tiles).
// out[M,1024] = A16[M,1024] x Wt16[1024 cols][1024]
// mode 0: Q -> qw16/qr16 (bias-added, [b][n][i][d]) + q32 raw
// mode 1: generic -> out0 [b][n][seq][d]   (K, KR)
// mode 3: V -> vT16 [b][n][d][seq] (transposed)
// ---------------------------------------------------------------------------
__global__ void proj_gemm(const _Float16* __restrict__ A, const _Float16* __restrict__ Wt,
                          int Mrows, int seqLen, int mode,
                          _Float16* __restrict__ out0, _Float16* __restrict__ out1,
                          float* __restrict__ q32,
                          const float* __restrict__ bias0, const float* __restrict__ bias1) {
  int wid  = (blockIdx.x * blockDim.x + threadIdx.x) >> 5;
  int lane = threadIdx.x & 31;
  int mtiles = Mrows >> 4;
  int mt = wid % mtiles;
  int ct = wid / mtiles;              // 64-wide column group
  if (ct >= (DM >> 6)) return;
  int li = lane & 15, half = lane >> 4;
  const _Float16* arow = A + (size_t)(mt * 16 + li) * DM;
  int colbase = ct * 64 + li;
  const _Float16* bcol = Wt + (size_t)colbase * DM;
  v8f acc[4] = {{}, {}, {}, {}};
#pragma unroll 4
  for (int s = 0; s < DM / 32; ++s) {
    v16h af = load_afrag(arow, s * 32, half);
#pragma unroll
    for (int t = 0; t < 4; ++t) {
      v16h bf = *(const v16h*)(bcol + (size_t)t * 16 * DM + s * 32 + 16 * half);
      acc[t] = wmma32(af, bf, acc[t]);
    }
  }
#pragma unroll
  for (int t = 0; t < 4; ++t) {
    int colg = colbase + t * 16;
    int n = colg >> 6, d = colg & 63;
    float b0 = bias0 ? bias0[colg] : 0.f;
    float b1 = bias1 ? bias1[colg] : 0.f;
#pragma unroll
    for (int r = 0; r < 8; ++r) {
      int mg = mt * 16 + r + 8 * half;
      int bb = mg % BSZ, seq = mg / BSZ;
      float v = acc[t][r];
      if (mode == 0) {
        size_t o = (((size_t)bb * NH + n) * QLEN + seq) * DH + d;
        out0[o] = (_Float16)(v + b0);
        out1[o] = (_Float16)(v + b1);
        q32[(size_t)mg * DM + colg] = v;
      } else if (mode == 3) {
        out0[(((size_t)bb * NH + n) * DH + d) * KLEN + seq] = (_Float16)v;
      } else {
        out0[(((size_t)bb * NH + n) * (size_t)seqLen + seq) * DH + d] = (_Float16)v;
      }
    }
  }
}

// ef[b][n][i][s] = sum_d (q[i,b,n,d] + r_s_bias[n,d]) * seg_embed[s,n,d]
__global__ void ef_kernel(const float* __restrict__ q32, const float* __restrict__ rsb,
                          const float* __restrict__ se, float* __restrict__ ef) {
  int tid = blockIdx.x * 256 + threadIdx.x;
  if (tid >= BSZ * NH * QLEN) return;
  int i = tid % QLEN;
  int n = (tid / QLEN) % NH;
  int b = tid / (QLEN * NH);
  const float* q  = q32 + ((size_t)i * BSZ + b) * DM + n * DH;
  const float* rb = rsb + n * DH;
  const float* s0 = se + n * DH;
  const float* s1 = se + NH * DH + n * DH;
  float e0 = 0.f, e1 = 0.f;
#pragma unroll 8
  for (int d = 0; d < DH; ++d) {
    float qv = q[d] + rb[d];
    e0 += qv * s0[d];
    e1 += qv * s1[d];
  }
  ef[(size_t)tid * 2 + 0] = e0;
  ef[(size_t)tid * 2 + 1] = e1;
}

// ---------------------------------------------------------------------------
// bd GEMM with fused rel_shift (per batch b), 16x64 output per wave:
// D[r,i] = sum_d kr[r,d]*qr[i,d]; store at j = r + i - QLEN -> BDT[n][j][i] f16
// ---------------------------------------------------------------------------
__global__ void bd_gemm(const _Float16* __restrict__ kr16, const _Float16* __restrict__ qr16,
                        _Float16* __restrict__ BDT, int b) {
  int wid  = (blockIdx.x * blockDim.x + threadIdx.x) >> 5;
  int lane = threadIdx.x & 31;
  int ig4 = wid & 15;                 // 16 groups of 4 i-tiles (QLEN/64)
  int rt  = (wid >> 4) % (RLEN / 16); // 192
  int n   = wid / (16 * (RLEN / 16));
  if (n >= NH) return;
  int li = lane & 15, half = lane >> 4;
  const _Float16* arow = kr16 + (((size_t)b * NH + n) * RLEN + rt * 16 + li) * DH;
  const _Float16* bcol = qr16 + (((size_t)b * NH + n) * QLEN + ig4 * 64 + li) * DH;
  v8f acc[4] = {{}, {}, {}, {}};
#pragma unroll
  for (int s = 0; s < 2; ++s) {
    v16h af = load_afrag(arow, s * 32, half);
#pragma unroll
    for (int t = 0; t < 4; ++t) {
      v16h bf = *(const v16h*)(bcol + (size_t)t * 16 * DH + s * 32 + 16 * half);
      acc[t] = wmma32(af, bf, acc[t]);
    }
  }
#pragma unroll
  for (int t = 0; t < 4; ++t) {
    int ig = ig4 * 64 + t * 16 + li;
#pragma unroll
    for (int r = 0; r < 8; ++r) {
      int rg = rt * 16 + r + 8 * half;
      int j  = rg + ig - QLEN;        // rel_shift: bd[i,j] = bd_full[i, j+QLEN-i]
      if ((unsigned)j < (unsigned)KLEN)
        BDT[((size_t)n * KLEN + j) * QLEN + ig] = (_Float16)acc[t][r];
    }
  }
}

// ---------------------------------------------------------------------------
// Fused attention (per batch b): one wave per (head, 16-row i-tile).
// Scores computed transposed: S^T[j,i] = K x Q^T so the D-layout maps directly
// onto the A-fragment of the P x V WMMA with no cross-lane traffic.
// Online softmax over j with causal mask (mask if j > MLEN+i).
// ---------------------------------------------------------------------------
__global__ void attn_kernel(const _Float16* __restrict__ k16, const _Float16* __restrict__ qw16,
                            const _Float16* __restrict__ vT16, const _Float16* __restrict__ BDT,
                            const float* __restrict__ ef, const unsigned char* __restrict__ sd8,
                            _Float16* __restrict__ vec16, int b) {
  int wid  = (blockIdx.x * blockDim.x + threadIdx.x) >> 5;
  int lane = threadIdx.x & 31;
  int it = wid & 63;
  int n  = wid >> 6;
  if (n >= NH) return;
  int li = lane & 15, half = lane >> 4;
  int icol = it * 16 + li;
  size_t hb = (size_t)b * NH + n;
  const _Float16* qrow = qw16 + (hb * QLEN + icol) * DH;
  v16h qf0 = *(const v16h*)(qrow + 16 * half);
  v16h qf1 = *(const v16h*)(qrow + 32 + 16 * half);
  float ef0 = ef[(hb * QLEN + icol) * 2 + 0];
  float ef1 = ef[(hb * QLEN + icol) * 2 + 1];
  const _Float16* kbase  = k16 + hb * KLEN * DH;
  const _Float16* vbase  = vT16 + hb * (size_t)DH * KLEN;
  const _Float16* bdbase = BDT + (size_t)n * KLEN * QLEN;
  const unsigned char* sdb = sd8 + (size_t)b * KLEN * QLEN;

  v8f O0 = {}, O1 = {}, O2 = {}, O3 = {};
  float m = -3.0e38f, l = 0.f;
  int jmax = MLEN + it * 16 + 15;   // last unmasked j for any row in this tile

  for (int j0 = 0; j0 <= jmax; j0 += 32) {
    // two transposed 16x16 score tiles over K=64 (2 wmma steps each)
    v8f s0 = {}, s1 = {};
    const _Float16* kr0 = kbase + (size_t)(j0 + li) * DH;
    const _Float16* kr1 = kr0 + 16 * DH;
    // prefetch next j-block's K rows toward the WGP (global_prefetch_b8)
    __builtin_prefetch(kr0 + 32 * DH, 0, 1);
    v16h a;
    a = load_afrag(kr0, 0, half);  s0 = wmma32(a, qf0, s0);
    a = load_afrag(kr0, 32, half); s0 = wmma32(a, qf1, s0);
    a = load_afrag(kr1, 0, half);  s1 = wmma32(a, qf0, s1);
    a = load_afrag(kr1, 32, half); s1 = wmma32(a, qf1, s1);

    float sc0[8], sc1[8];
    float tmax = -3.0e38f;
#pragma unroll
    for (int r = 0; r < 8; ++r) {
      int j = j0 + r + 8 * half;
      float bd  = (float)bdbase[(size_t)j * QLEN + icol];
      float efv = sdb[(size_t)j * QLEN + icol] ? ef1 : ef0;
      float v = (s0[r] + bd + efv) * SCALE;
      if (j > MLEN + icol) v -= 1.0e30f;
      sc0[r] = v; tmax = fmaxf(tmax, v);
      int j2 = j + 16;
      bd  = (float)bdbase[(size_t)j2 * QLEN + icol];
      efv = sdb[(size_t)j2 * QLEN + icol] ? ef1 : ef0;
      v = (s1[r] + bd + efv) * SCALE;
      if (j2 > MLEN + icol) v -= 1.0e30f;
      sc1[r] = v; tmax = fmaxf(tmax, v);
    }
    tmax = fmaxf(tmax, __shfl_xor(tmax, 16));
    float mnew  = fmaxf(m, tmax);
    float alpha = __expf(m - mnew);
    float psum = 0.f;
    v16h pf;
#pragma unroll
    for (int r = 0; r < 8; ++r) {
      float p0 = __expf(sc0[r] - mnew);
      float p1 = __expf(sc1[r] - mnew);
      psum += p0 + p1;
      pf[r]     = (_Float16)p0;   // A-frag elems 0..7  : j = j0 + 8h + r
      pf[r + 8] = (_Float16)p1;   // A-frag elems 8..15 : j = j0 + 16 + 8h + r
    }
    psum += __shfl_xor(psum, 16);
    l = l * alpha + psum;
    m = mnew;
    // rescale accumulator rows (O rows = r + 8*half; stats live in lane == row)
#pragma unroll
    for (int r = 0; r < 8; ++r) {
      float av = __shfl(alpha, r + 8 * half);
      O0[r] *= av; O1[r] *= av; O2[r] *= av; O3[r] *= av;
    }
    const _Float16* vcol = vbase + (size_t)li * KLEN + j0 + 8 * half;
    v16h bv0 = __builtin_shufflevector(*(const v8h*)(vcol), *(const v8h*)(vcol + 16), 0, 1, 2, 3, 4, 5, 6, 7, 8, 9, 10, 11, 12, 13, 14, 15);
    v16h bv1 = __builtin_shufflevector(*(const v8h*)(vcol + (size_t)16 * KLEN), *(const v8h*)(vcol + (size_t)16 * KLEN + 16), 0, 1, 2, 3, 4, 5, 6, 7, 8, 9, 10, 11, 12, 13, 14, 15);
    v16h bv2 = __builtin_shufflevector(*(const v8h*)(vcol + (size_t)32 * KLEN), *(const v8h*)(vcol + (size_t)32 * KLEN + 16), 0, 1, 2, 3, 4, 5, 6, 7, 8, 9, 10, 11, 12, 13, 14, 15);
    v16h bv3 = __builtin_shufflevector(*(const v8h*)(vcol + (size_t)48 * KLEN), *(const v8h*)(vcol + (size_t)48 * KLEN + 16), 0, 1, 2, 3, 4, 5, 6, 7, 8, 9, 10, 11, 12, 13, 14, 15);
    O0 = wmma32(pf, bv0, O0);
    O1 = wmma32(pf, bv1, O1);
    O2 = wmma32(pf, bv2, O2);
    O3 = wmma32(pf, bv3, O3);
  }
  // finalize: divide by l, store attn_vec f16 at [(i*B+b)][n*64+d]
#pragma unroll
  for (int r = 0; r < 8; ++r) {
    float lv  = __shfl(l, r + 8 * half);
    float inv = 1.0f / lv;
    int ig = it * 16 + r + 8 * half;
    _Float16* o = vec16 + ((size_t)ig * BSZ + b) * DM + n * DH;
    o[0 * 16 + li] = (_Float16)(O0[r] * inv);
    o[1 * 16 + li] = (_Float16)(O1[r] * inv);
    o[2 * 16 + li] = (_Float16)(O2[r] * inv);
    o[3 * 16 + li] = (_Float16)(O3[r] * inv);
  }
}

// ---------------------------------------------------------------------------
// Output projection, 16x64 per wave: ao[m,h] = vec16[m,:] . wo16[h,:] + h_in[m,h]
// (wo is [h][n][d] row-major == [col][k] layout already)
// ---------------------------------------------------------------------------
__global__ void out_gemm(const _Float16* __restrict__ vec16, const _Float16* __restrict__ wo16,
                         const float* __restrict__ hin, float* __restrict__ ao) {
  int wid  = (blockIdx.x * blockDim.x + threadIdx.x) >> 5;
  int lane = threadIdx.x & 31;
  int mtiles = (QLEN * BSZ) >> 4;
  int mt = wid % mtiles;
  int ct = wid / mtiles;
  if (ct >= (DM >> 6)) return;
  int li = lane & 15, half = lane >> 4;
  const _Float16* arow = vec16 + (size_t)(mt * 16 + li) * DM;
  int colbase = ct * 64 + li;
  const _Float16* bcol = wo16 + (size_t)colbase * DM;
  v8f acc[4] = {{}, {}, {}, {}};
#pragma unroll 4
  for (int s = 0; s < DM / 32; ++s) {
    v16h af = load_afrag(arow, s * 32, half);
#pragma unroll
    for (int t = 0; t < 4; ++t) {
      v16h bf = *(const v16h*)(bcol + (size_t)t * 16 * DM + s * 32 + 16 * half);
      acc[t] = wmma32(af, bf, acc[t]);
    }
  }
#pragma unroll
  for (int t = 0; t < 4; ++t) {
    int colg = colbase + t * 16;
#pragma unroll
    for (int r = 0; r < 8; ++r) {
      int mg = mt * 16 + r + 8 * half;
      ao[(size_t)mg * DM + colg] = acc[t][r] + hin[(size_t)mg * DM + colg];
    }
  }
}

// ---------------------------------------------------------------------------
// LayerNorm over last dim, one block per row
// ---------------------------------------------------------------------------
__global__ void ln_kernel(const float* __restrict__ ao, const float* __restrict__ gamma,
                          const float* __restrict__ beta, float* __restrict__ out) {
  __shared__ float s1[256], s2[256];
  int row = blockIdx.x, t = threadIdx.x;
  const float* x = ao + (size_t)row * DM;
  float a = 0.f, b = 0.f;
  for (int c = t; c < DM; c += 256) { float v = x[c]; a += v; b += v * v; }
  s1[t] = a; s2[t] = b;
  __syncthreads();
  for (int o = 128; o > 0; o >>= 1) {
    if (t < o) { s1[t] += s1[t + o]; s2[t] += s2[t + o]; }
    __syncthreads();
  }
  float mu  = s1[0] / DM;
  float var = s2[0] / DM - mu * mu;
  float rs  = rsqrtf(var + LN_EPS);
  for (int c = t; c < DM; c += 256)
    out[(size_t)row * DM + c] = (x[c] - mu) * rs * gamma[c] + beta[c];
}

// ---------------------------------------------------------------------------
extern "C" void kernel_launch(void* const* d_in, const int* in_sizes, int n_in,
                              void* d_out, int out_size, void* d_ws, size_t ws_size,
                              hipStream_t stream) {
  const float* h_in    = (const float*)d_in[0];
  const float* mems    = (const float*)d_in[1];
  const float* r_in    = (const float*)d_in[2];
  const float* seg_mat = (const float*)d_in[3];
  // d_in[4] = attn_mask (causal, recomputed analytically)
  const float* wq = (const float*)d_in[5];
  const float* wk = (const float*)d_in[6];
  const float* wv = (const float*)d_in[7];
  const float* wo = (const float*)d_in[8];
  const float* wr = (const float*)d_in[9];
  const float* rwb = (const float*)d_in[10];
  const float* rrb = (const float*)d_in[11];
  const float* rsb = (const float*)d_in[12];
  const float* se  = (const float*)d_in[13];
  const float* gam = (const float*)d_in[14];
  const float* bet = (const float*)d_in[15];
  float* out = (float*)d_out;

  char* p = (char*)d_ws;
  auto alloc = [&](size_t bytes) -> char* {
    char* r0 = p; p += (bytes + 255) & ~(size_t)255; return r0;
  };
  _Float16* cat16 = (_Float16*)alloc((size_t)KLEN * BSZ * DM * 2);
  _Float16* r16   = (_Float16*)alloc((size_t)RLEN * BSZ * DM * 2);
  _Float16* wqT   = (_Float16*)alloc((size_t)DM * DM * 2);
  _Float16* wkT   = (_Float16*)alloc((size_t)DM * DM * 2);
  _Float16* wvT   = (_Float16*)alloc((size_t)DM * DM * 2);
  _Float16* wrT   = (_Float16*)alloc((size_t)DM * DM * 2);
  _Float16* wo16  = (_Float16*)alloc((size_t)DM * DM * 2);
  _Float16* qw16  = (_Float16*)alloc((size_t)BSZ * NH * QLEN * DH * 2);
  _Float16* qr16  = (_Float16*)alloc((size_t)BSZ * NH * QLEN * DH * 2);
  float*    q32   = (float*)alloc((size_t)QLEN * BSZ * DM * 4);
  _Float16* k16   = (_Float16*)alloc((size_t)BSZ * NH * KLEN * DH * 2);
  _Float16* kr16  = (_Float16*)alloc((size_t)BSZ * NH * RLEN * DH * 2);
  _Float16* vT16  = (_Float16*)alloc((size_t)BSZ * NH * DH * KLEN * 2);
  float*    efb   = (float*)alloc((size_t)BSZ * NH * QLEN * 2 * 4);
  unsigned char* sd8 = (unsigned char*)alloc((size_t)BSZ * KLEN * QLEN);
  _Float16* vec16 = (_Float16*)alloc((size_t)QLEN * BSZ * DM * 2);
  float*    ao32  = (float*)alloc((size_t)QLEN * BSZ * DM * 4);
  _Float16* BDT   = (_Float16*)alloc((size_t)NH * KLEN * QLEN * 2);
  (void)ws_size; (void)n_in; (void)in_sizes; (void)out_size;

  // staging
  conv_cat<<<(KLEN * BSZ * DM + 255) / 256, 256, 0, stream>>>(mems, h_in, cat16);
  conv_f16<<<(RLEN * BSZ * DM + 255) / 256, 256, 0, stream>>>(r_in, r16, (size_t)RLEN * BSZ * DM);
  conv_wT<<<(DM * DM + 255) / 256, 256, 0, stream>>>(wq, wqT);
  conv_wT<<<(DM * DM + 255) / 256, 256, 0, stream>>>(wk, wkT);
  conv_wT<<<(DM * DM + 255) / 256, 256, 0, stream>>>(wv, wvT);
  conv_wT<<<(DM * DM + 255) / 256, 256, 0, stream>>>(wr, wrT);
  conv_f16<<<(DM * DM + 255) / 256, 256, 0, stream>>>(wo, wo16, (size_t)DM * DM);
  sd8_kernel<<<(BSZ * KLEN * QLEN + 255) / 256, 256, 0, stream>>>(seg_mat, sd8);

  // projections (WMMA GEMMs, 16x64 per wave)
  {
    int waves = ((QLEN * BSZ) / 16) * (DM / 64);
    proj_gemm<<<waves / 8, 256, 0, stream>>>(cat16 + (size_t)MLEN * BSZ * DM, wqT,
                                             QLEN * BSZ, QLEN, 0, qw16, qr16, q32, rwb, rrb);
  }
  {
    int waves = ((KLEN * BSZ) / 16) * (DM / 64);
    proj_gemm<<<waves / 8, 256, 0, stream>>>(cat16, wkT, KLEN * BSZ, KLEN, 1, k16,
                                             nullptr, nullptr, nullptr, nullptr);
    proj_gemm<<<waves / 8, 256, 0, stream>>>(cat16, wvT, KLEN * BSZ, KLEN, 3, vT16,
                                             nullptr, nullptr, nullptr, nullptr);
  }
  {
    int waves = ((RLEN * BSZ) / 16) * (DM / 64);
    proj_gemm<<<waves / 8, 256, 0, stream>>>(r16, wrT, RLEN * BSZ, RLEN, 1, kr16,
                                             nullptr, nullptr, nullptr, nullptr);
  }
  ef_kernel<<<(BSZ * NH * QLEN) / 256, 256, 0, stream>>>(q32, rsb, se, efb);

  // per-batch: bd GEMM (+rel_shift) into L2-resident scratch, then fused attention
  for (int b = 0; b < BSZ; ++b) {
    int bdw = NH * (RLEN / 16) * (QLEN / 64);
    bd_gemm<<<bdw / 8, 256, 0, stream>>>(kr16, qr16, BDT, b);
    attn_kernel<<<(NH * 64) / 4, 128, 0, stream>>>(k16, qw16, vT16, BDT, efb, sd8,
                                                   vec16, b);
  }

  // output projection + residual, then layernorm
  {
    int waves = ((QLEN * BSZ) / 16) * (DM / 64);
    out_gemm<<<waves / 8, 256, 0, stream>>>(vec16, wo16, h_in, ao32);
  }
  ln_kernel<<<QLEN * BSZ, 256, 0, stream>>>(ao32, gam, bet, out);
}
